// OldMultiHeadAttention_41154376631033
// MI455X (gfx1250) — matrix operand order repaired and checked by reading, not compiled, not run on hardware
//
#include <hip/hip_runtime.h>
#include <hip/hip_bf16.h>

// ---------------------------------------------------------------------------
// MultiHeadAttention forward for MI455X (gfx1250, wave32, WMMA bf16).
// B=2, S=2048, D=768, H=12, Dh=64.
// Pipeline: cvt/transpose -> QKV GEMM (wmma) -> flash attention (wmma) ->
//           output projection GEMM (wmma).
// ---------------------------------------------------------------------------

typedef __attribute__((ext_vector_type(16))) __bf16 v16bf;
typedef __attribute__((ext_vector_type(8)))  float v8f;

#define BATCH 2
#define SEQ   2048
#define DMODEL 768
#define NHEAD 12
#define DHEAD 64
#define NROWS (BATCH * SEQ)          // 4096
#define NQKV  (3 * DMODEL)           // 2304

union Frag16 {                       // 32 bytes = one 16-bf16 WMMA operand
    v16bf v;
    uint4 q[2];
    unsigned short h[16];
};

__device__ __forceinline__ unsigned short f32_to_bf16(float f) {
    union { float f; unsigned int u; } cv; cv.f = f;
    unsigned int u = cv.u;
    u += 0x7FFFu + ((u >> 16) & 1u);         // round-to-nearest-even
    return (unsigned short)(u >> 16);
}

// In-row (16-lane) butterfly reductions via DPP16 (pure VALU, no LDS).
// DPP rows are 16 lanes on wave32 -- exactly the WMMA C-layout row groups.
#define DPP_F32(x, ctrl)                                                    \
    __builtin_bit_cast(float, __builtin_amdgcn_mov_dpp(                     \
        __builtin_bit_cast(int, (x)), (ctrl), 0xF, 0xF, true))

__device__ __forceinline__ float row16_max(float t) {
    t = fmaxf(t, DPP_F32(t, 0xB1));   // quad_perm [1,0,3,2]
    t = fmaxf(t, DPP_F32(t, 0x4E));   // quad_perm [2,3,0,1]
    t = fmaxf(t, DPP_F32(t, 0x141));  // row_half_mirror (cross-quad)
    t = fmaxf(t, DPP_F32(t, 0x140));  // row_mirror (cross-half)
    return t;
}
__device__ __forceinline__ float row16_sum(float t) {
    t += DPP_F32(t, 0xB1);
    t += DPP_F32(t, 0x4E);
    t += DPP_F32(t, 0x141);
    t += DPP_F32(t, 0x140);
    return t;
}

// ---------------------------------------------------------------------------
// 1a. fp32 -> bf16 row-major
// ---------------------------------------------------------------------------
__global__ void k_cvt(const float* __restrict__ src,
                      unsigned short* __restrict__ dst, int n) {
    int i = blockIdx.x * blockDim.x + threadIdx.x;
    if (i < n) dst[i] = f32_to_bf16(src[i]);
}

// ---------------------------------------------------------------------------
// 1b. W (K x N fp32) -> Wt (N x K bf16) so B-fragments load as two contiguous 16B runs.
// ---------------------------------------------------------------------------
__global__ void k_transpose_w(const float* __restrict__ w,
                              unsigned short* __restrict__ wt, int K, int N) {
    int i = blockIdx.x * blockDim.x + threadIdx.x;
    if (i >= K * N) return;
    int k = i / N;
    int n = i - k * N;
    wt[(size_t)n * K + k] = f32_to_bf16(w[i]);
}

// ---------------------------------------------------------------------------
// 1c. int mask -> additive float bias (0 / -inf), one v_add in the hot loop.
// ---------------------------------------------------------------------------
__global__ void k_maskbias(const int* __restrict__ mask,
                           float* __restrict__ mb, int n) {
    int i = blockIdx.x * blockDim.x + threadIdx.x;
    if (i < n) mb[i] = mask[i] ? 0.0f : -__builtin_inff();
}

// ---------------------------------------------------------------------------
// 2. Tiled bf16 WMMA GEMM:  C[M,N] = A[M,K] * Bt[N,K]^T + bias
//    Block = 256 threads (8 waves); wave computes 32(M) x 64(N): every
//    B fragment feeds two v_wmma (two M-subtiles) -> 2x arithmetic intensity.
//    MODE 0: fp32 row-major out.  MODE 1: QKV scatter epilogue.
// ---------------------------------------------------------------------------
template <int MODE>
__global__ __launch_bounds__(256, 1)
void k_gemm_bf16(const unsigned short* __restrict__ A,   // M x K
                 const unsigned short* __restrict__ Bt,  // N x K
                 const float* __restrict__ bias,         // N
                 float* __restrict__ Cout,
                 unsigned short* __restrict__ Qb,
                 unsigned short* __restrict__ Kb,
                 unsigned short* __restrict__ Vt,
                 int M, int N, int K) {
    const int wave = threadIdx.x >> 5;
    const int lane = threadIdx.x & 31;
    const int r  = lane & 15;        // row/col within 16
    const int kh = lane >> 4;        // K-half select (wave32 operand layout)

    const int row0 = blockIdx.y * 256 + wave * 32;
    const int n0   = blockIdx.x * 64;

    v8f zero = {};
    v8f acc[8] = {zero, zero, zero, zero, zero, zero, zero, zero};

    for (int kb = 0; kb < K; kb += 32) {
        // A fragments (16x32 bf16): lane<16 -> K = kb+{0..7,16..23},
        //                           lane>=16 -> K = kb+{8..15,24..31}
        Frag16 fa0, fa1;
        {
            const uint4* pa = reinterpret_cast<const uint4*>(
                A + (size_t)(row0 + r) * K + kb + kh * 8);
            fa0.q[0] = pa[0]; fa0.q[1] = pa[2];
            const uint4* pb = reinterpret_cast<const uint4*>(
                A + (size_t)(row0 + 16 + r) * K + kb + kh * 8);
            fa1.q[0] = pb[0]; fa1.q[1] = pb[2];
        }
#pragma unroll
        for (int j = 0; j < 4; ++j) {
            // B fragment (32x16): lane holds column n0+j*16+r,
            // K = kb + kh*8 + {0..7,16..23} (same order as A).
            Frag16 fb;
            const uint4* pb = reinterpret_cast<const uint4*>(
                Bt + (size_t)(n0 + j * 16 + r) * K + kb + kh * 8);
            fb.q[0] = pb[0];
            fb.q[1] = pb[2];
            acc[j] = __builtin_amdgcn_wmma_f32_16x16x32_bf16(
                false, fa0.v, false, fb.v, (short)0, acc[j], false, false);
            acc[4 + j] = __builtin_amdgcn_wmma_f32_16x16x32_bf16(
                false, fa1.v, false, fb.v, (short)0, acc[4 + j], false, false);
        }
    }

    // C layout: lanes 0-15 cols, VGPR g = row g (low lanes) / g+8 (high lanes)
#pragma unroll
    for (int mi = 0; mi < 2; ++mi)
#pragma unroll
        for (int j = 0; j < 4; ++j) {
            const int col = n0 + j * 16 + r;
            const float bsv = bias[col];
#pragma unroll
            for (int g = 0; g < 8; ++g) {
                const int row = row0 + mi * 16 + g + kh * 8;
                const float v = acc[mi * 4 + j][g] + bsv;
                if (MODE == 0) {
                    Cout[(size_t)row * N + col] = v;
                } else {
                    const int b = row >> 11, s = row & 2047;
                    const int sel = col / DMODEL;
                    const int cc  = col - sel * DMODEL;
                    const int hh  = cc >> 6, d = cc & 63;
                    const size_t bh = (size_t)(b * NHEAD + hh);
                    if (sel == 0)       // Q: fold in softmax scale 1/sqrt(64)
                        Qb[(bh * SEQ + s) * DHEAD + d] =
                            f32_to_bf16(v * 0.125f);
                    else if (sel == 1)  // K: head-major rows
                        Kb[(bh * SEQ + s) * DHEAD + d] = f32_to_bf16(v);
                    else                // V: transposed (B,H,Dh,S)
                        Vt[(bh * DHEAD + d) * SEQ + s] = f32_to_bf16(v);
                }
            }
        }
}

// ---------------------------------------------------------------------------
// 3. Flash attention tile body (32 keys). CAUSAL only for the diagonal tile.
// ---------------------------------------------------------------------------
template <bool CAUSAL>
__device__ __forceinline__ void attn_tile(
    int key0, int q0, int r, int kh,
    const unsigned short* __restrict__ Kbh,   // K for this (b,h): S x Dh
    const unsigned short* __restrict__ Vth,   // V^T for this (b,h): Dh x S
    const float* __restrict__ mbp,            // additive mask bias for b
    const Frag16& aq0, const Frag16& aq1,
    float* m_i, float* l_i, v8f* acc, unsigned short* pl) {
    v8f sc[2];
#pragma unroll
    for (int ch = 0; ch < 2; ++ch) {
        const int keyc = key0 + ch * 16 + r;   // this lane's score column
        const uint4* pk = reinterpret_cast<const uint4*>(
            Kbh + (size_t)keyc * DHEAD + kh * 8);
        Frag16 bk0, bk1;
        bk0.q[0] = pk[0]; bk0.q[1] = pk[2];    // K-dim 0..31
        bk1.q[0] = pk[4]; bk1.q[1] = pk[6];    // K-dim 32..63
        v8f z = {};
        v8f s = __builtin_amdgcn_wmma_f32_16x16x32_bf16(
            false, aq0.v, false, bk0.v, (short)0, z, false, false);
        s = __builtin_amdgcn_wmma_f32_16x16x32_bf16(
            false, aq1.v, false, bk1.v, (short)0, s, false, false);
        const float mbv = mbp[keyc];           // 0 or -inf
#pragma unroll
        for (int g = 0; g < 8; ++g) {
            s[g] += mbv;
            if (CAUSAL) {
                const int qi = q0 + g + kh * 8;
                if (keyc > qi) s[g] = -__builtin_inff();
            }
        }
        sc[ch] = s;
    }

    // Online softmax stats (row = g + kh*8; DPP16 butterflies == row groups).
    // exp args clamped at -80 so (-inf)-(-inf)=NaN folds to exp(-80)=~0
    // through v_max_num NaN semantics (no compares/selects needed).
#pragma unroll
    for (int g = 0; g < 8; ++g) {
        const float t = row16_max(fmaxf(sc[0][g], sc[1][g]));
        const float mn = fmaxf(m_i[g], t);
        const float alpha = __expf(fmaxf(m_i[g] - mn, -80.0f));
        const float p0 = __expf(fmaxf(sc[0][g] - mn, -80.0f));
        const float p1 = __expf(fmaxf(sc[1][g] - mn, -80.0f));
        sc[0][g] = p0; sc[1][g] = p1;
        l_i[g] = l_i[g] * alpha + row16_sum(p0 + p1);
        m_i[g] = mn;
#pragma unroll
        for (int j = 0; j < 4; ++j) acc[j][g] *= alpha;
    }

    // transpose P: C-layout regs -> LDS -> A-layout fragment
#pragma unroll
    for (int ch = 0; ch < 2; ++ch)
#pragma unroll
        for (int g = 0; g < 8; ++g)
            pl[(g + kh * 8) * 32 + ch * 16 + r] = f32_to_bf16(sc[ch][g]);
    asm volatile("s_wait_dscnt 0" ::: "memory");
    Frag16 ap;
    const uint4* pp = reinterpret_cast<const uint4*>(pl + r * 32 + kh * 8);
    ap.q[0] = pp[0]; ap.q[1] = pp[2];

    // O(16x64) += P(16x32) * V(32x64); Vt rows give contiguous B-frags
#pragma unroll
    for (int j = 0; j < 4; ++j) {
        const int dh = j * 16 + r;
        const uint4* pv = reinterpret_cast<const uint4*>(
            Vth + (size_t)dh * SEQ + key0 + kh * 8);
        Frag16 bv;
        bv.q[0] = pv[0]; bv.q[1] = pv[2];
        acc[j] = __builtin_amdgcn_wmma_f32_16x16x32_bf16(
            false, ap.v, false, bv.v, (short)0, acc[j], false, false);
    }
}

// Grid (S/16/4, B*H), block 128 (4 waves); each wave owns one 16-query tile.
__global__ __launch_bounds__(128, 1)
void k_attn(const unsigned short* __restrict__ Qb,
            const unsigned short* __restrict__ Kb,
            const unsigned short* __restrict__ Vt,
            const float* __restrict__ maskbias,
            unsigned short* __restrict__ vwb) {
    __shared__ unsigned short lds_p[4][16 * 32];   // per-wave P tile (bf16)

    const int wave = threadIdx.x >> 5;
    const int lane = threadIdx.x & 31;
    const int r  = lane & 15;
    const int kh = lane >> 4;

    const int bh = blockIdx.y;               // 0..23
    const int b  = bh / NHEAD;
    const int h  = bh - b * NHEAD;
    const int qt = blockIdx.x * 4 + wave;    // 0..127
    const int q0 = qt * 16;

    const unsigned short* Kbh = Kb + (size_t)bh * SEQ * DHEAD;
    const unsigned short* Vth = Vt + (size_t)bh * DHEAD * SEQ;
    const float* mbp = maskbias + b * SEQ;

    // Q fragments for K-dim 0..31 and 32..63 (A-operand layout)
    const unsigned short* Qrow = Qb + ((size_t)bh * SEQ + q0) * DHEAD;
    Frag16 aq0, aq1;
    {
        const uint4* p0 = reinterpret_cast<const uint4*>(
            Qrow + (size_t)r * DHEAD + kh * 8);
        aq0.q[0] = p0[0]; aq0.q[1] = p0[2];
        const uint4* p1 = reinterpret_cast<const uint4*>(
            Qrow + (size_t)r * DHEAD + 32 + kh * 8);
        aq1.q[0] = p1[0]; aq1.q[1] = p1[2];
    }

    float m_i[8], l_i[8];
#pragma unroll
    for (int g = 0; g < 8; ++g) { m_i[g] = -__builtin_inff(); l_i[g] = 0.0f; }
    v8f zero = {};
    v8f acc[4] = {zero, zero, zero, zero};
    unsigned short* pl = lds_p[wave];

    // All tiles before the diagonal satisfy key <= q unconditionally.
    const int ktmax = (q0 + 15) >> 5;
    for (int kt = 0; kt < ktmax; ++kt)
        attn_tile<false>(kt * 32, q0, r, kh, Kbh, Vth, mbp,
                         aq0, aq1, m_i, l_i, acc, pl);
    attn_tile<true>(ktmax * 32, q0, r, kh, Kbh, Vth, mbp,
                    aq0, aq1, m_i, l_i, acc, pl);

    // normalize and store bf16 vw (token-major B*S x D for the last GEMM)
#pragma unroll
    for (int g = 0; g < 8; ++g) {
        const float inv = (l_i[g] > 0.0f) ? 1.0f / l_i[g] : 0.0f;
        const int m = g + kh * 8;
        const size_t base = (size_t)(b * SEQ + q0 + m) * DMODEL + h * DHEAD;
#pragma unroll
        for (int j = 0; j < 4; ++j)
            vwb[base + j * 16 + r] = f32_to_bf16(acc[j][g] * inv);
    }
}

// ---------------------------------------------------------------------------
// Host-side launcher
// ---------------------------------------------------------------------------
extern "C" void kernel_launch(void* const* d_in, const int* in_sizes, int n_in,
                              void* d_out, int out_size, void* d_ws,
                              size_t ws_size, hipStream_t stream) {
    const float* x     = (const float*)d_in[0];
    const float* W_qkv = (const float*)d_in[1];
    const float* b_qkv = (const float*)d_in[2];
    const float* W_vw  = (const float*)d_in[3];
    const float* b_vw  = (const float*)d_in[4];
    const int*   mask  = (const int*)d_in[5];

    char* ws = (char*)d_ws;
    unsigned short* Xbf   = (unsigned short*)ws; ws += (size_t)NROWS * DMODEL * 2;
    unsigned short* WqkvT = (unsigned short*)ws; ws += (size_t)NQKV * DMODEL * 2;
    unsigned short* WvwT  = (unsigned short*)ws; ws += (size_t)DMODEL * DMODEL * 2;
    unsigned short* Qb    = (unsigned short*)ws; ws += (size_t)BATCH * NHEAD * SEQ * DHEAD * 2;
    unsigned short* Kb    = (unsigned short*)ws; ws += (size_t)BATCH * NHEAD * SEQ * DHEAD * 2;
    unsigned short* Vt    = (unsigned short*)ws; ws += (size_t)BATCH * NHEAD * SEQ * DHEAD * 2;
    unsigned short* vwb   = (unsigned short*)ws; ws += (size_t)NROWS * DMODEL * 2;
    float*          mb    = (float*)ws;          ws += (size_t)BATCH * SEQ * 4;

    // 1) precision conversion / weight transposition / mask bias
    {
        int n = NROWS * DMODEL;
        k_cvt<<<(n + 255) / 256, 256, 0, stream>>>(x, Xbf, n);
        int nq = DMODEL * NQKV;
        k_transpose_w<<<(nq + 255) / 256, 256, 0, stream>>>(W_qkv, WqkvT,
                                                            DMODEL, NQKV);
        int nv = DMODEL * DMODEL;
        k_transpose_w<<<(nv + 255) / 256, 256, 0, stream>>>(W_vw, WvwT,
                                                            DMODEL, DMODEL);
        int nm = BATCH * SEQ;
        k_maskbias<<<(nm + 255) / 256, 256, 0, stream>>>(mask, mb, nm);
    }

    // 2) QKV projection (WMMA) with head-major scatter epilogue
    {
        dim3 grid(NQKV / 64, NROWS / 256);   // 36 x 16
        k_gemm_bf16<1><<<grid, 256, 0, stream>>>(Xbf, WqkvT, b_qkv, nullptr,
                                                 Qb, Kb, Vt,
                                                 NROWS, NQKV, DMODEL);
    }

    // 3) causal flash attention (WMMA)
    {
        dim3 grid(SEQ / 16 / 4, BATCH * NHEAD);  // 32 x 24
        k_attn<<<grid, 128, 0, stream>>>(Qb, Kb, Vt, mb, vwb);
    }

    // 4) output projection (WMMA) -> fp32 d_out
    {
        dim3 grid(DMODEL / 64, NROWS / 256);  // 12 x 16
        k_gemm_bf16<0><<<grid, 256, 0, stream>>>(vwb, WvwT, b_vw,
                                                 (float*)d_out,
                                                 nullptr, nullptr, nullptr,
                                                 NROWS, DMODEL, DMODEL);
    }
}
